// CSBrain_6004364279926
// MI455X (gfx1250) — matrix operand order repaired and checked by reading, not compiled, not run on hardware
//
#include <hip/hip_runtime.h>
#include <cstdint>
#include <cstddef>

// ---------------- types ----------------
typedef __bf16 bf16;
typedef __attribute__((ext_vector_type(16))) __bf16 bf16x16;
typedef __attribute__((ext_vector_type(8)))  __bf16 bf16x8;
typedef __attribute__((ext_vector_type(8)))  float  f32x8;

// ---------------- model constants ----------------
#define BB      32
#define CC      19
#define TT      128
#define DMOD    200
#define DIN     200
#define NTOK    77824      // BB*CC*TT
#define KP      224        // DMOD padded to mult of 32
#define DFF     800
#define NG      20480      // BB*5*TT  (group-mean rows)
#define LDQKV   768        // 3 chunks * 4 heads * 64 (head dim 50 padded to 64)
#define NHEAD   4
#define DH      50
#define SCALE_ATTN 0.14142135623730951f   // 1/sqrt(50)

// ---------------- device helpers ----------------
__device__ inline float bf2f(bf16 b){
  unsigned short h = __builtin_bit_cast(unsigned short, b);
  unsigned u = ((unsigned)h) << 16;
  return __builtin_bit_cast(float, u);
}
__device__ inline bf16 f2bf(float f){
  unsigned u = __builtin_bit_cast(unsigned, f);
  unsigned r = u + 0x7FFFu + ((u >> 16) & 1u);
  unsigned short h = (unsigned short)(r >> 16);
  return __builtin_bit_cast(bf16, h);
}
__device__ inline bf16 bfzero(){ return __builtin_bit_cast(bf16, (unsigned short)0); }
__device__ inline f32x8 fz8(){ f32x8 z;
#pragma unroll
  for (int i=0;i<8;i++) z[i]=0.f;
  return z; }

// A-matrix fragment (16x32 bf16): lane holds row M=lane&15; K chunks
// [hi*8, hi*8+8) and [16+hi*8, 16+hi*8+8)  (CDNA5 ISA 7.12.2)
__device__ inline bf16x16 load_a(const bf16* p, int hi){
  bf16x8 c0 = *(const bf16x8*)(p + hi*8);
  bf16x8 c1 = *(const bf16x8*)(p + 16 + hi*8);
  bf16x16 r;
#pragma unroll
  for (int i=0;i<8;i++){ r[i]=c0[i]; r[i+8]=c1[i]; }
  return r;
}
// B-matrix fragment (32x16 bf16): lane holds column N=lane&15 (= a row of W);
// K chunks [hi*8, hi*8+8) and [16+hi*8, 16+hi*8+8) (same as A).
__device__ inline bf16x16 load_b(const bf16* p, int hi){
  bf16x8 c0 = *(const bf16x8*)(p + hi*8);
  bf16x8 c1 = *(const bf16x8*)(p + 16 + hi*8);
  bf16x16 r;
#pragma unroll
  for (int i=0;i<8;i++){ r[i]=c0[i]; r[i+8]=c1[i]; }
  return r;
}
__device__ inline f32x8 wmma_bf(bf16x16 a, bf16x16 b, f32x8 c){
  return __builtin_amdgcn_wmma_f32_16x16x32_bf16(false, a, false, b, (short)0, c, false, false);
}

// ---------------- weight fp32 -> padded bf16 ----------------
__global__ void cvt_w_kernel(const float* __restrict__ src, int rows, int cols,
                             bf16* __restrict__ dst, int ldd){
  int i = blockIdx.x * 256 + threadIdx.x;
  if (i < rows * cols){
    int r = i / cols, c = i - r * cols;
    dst[(size_t)r * ldd + c] = f2bf(src[i]);
  }
}

// ---------------- generic WMMA GEMM ----------------
// out[m, n] = sum_k A[m,k] * W[n,k] + bias[n]
// MODE 0: float out, accumulate (+=)       (residual adds)
// MODE 1: bf16 out, identity columns
// MODE 2: bf16 out, ReLU
// MODE 3: bf16 out, qkv head-pad column map into LDQKV-stride rows
template<int MODE>
__global__ __launch_bounds__(128) void gemm_wmma_k(
    const bf16* __restrict__ A, int lda, int kIters,
    const bf16* __restrict__ W, int ldw,
    const float* __restrict__ bias, int Kout,
    void* __restrict__ outp, int ldc)
{
  const int w = threadIdx.x >> 5, lane = threadIdx.x & 31;
  const int hi = lane >> 4, lm = lane & 15;
  const int m0 = blockIdx.x * 16;
  const int ntile0 = (blockIdx.y * 4 + w) * 4;
  const int KoutT = (Kout + 15) >> 4;
  bool act[4]; bool any = false;
#pragma unroll
  for (int j=0;j<4;j++){ act[j] = (ntile0 + j) < KoutT; any = any || act[j]; }
  if (!any) return;
  const bf16* arow = A + (size_t)(m0 + lm) * lda;
  f32x8 acc[4] = { fz8(), fz8(), fz8(), fz8() };
  for (int kt = 0; kt < kIters; ++kt){
    const int k = kt * 32;
    bf16x16 af = load_a(arow + k, hi);
#pragma unroll
    for (int j=0;j<4;j++){
      if (!act[j]) continue;
      const bf16* wrow = W + (size_t)((ntile0+j)*16 + lm) * ldw + k;
      acc[j] = wmma_bf(af, load_b(wrow, hi), acc[j]);
    }
  }
#pragma unroll
  for (int j=0;j<4;j++){
    if (!act[j]) continue;
    int col = (ntile0+j)*16 + lm;
    if (col >= Kout) continue;
    float bv = bias[col];
    int oc = col;
    if (MODE == 3){
      int chunk = col / 200; int rem = col - chunk*200;
      int h = rem / 50; int dh = rem - h*50;
      oc = chunk*256 + h*64 + dh;
    }
#pragma unroll
    for (int r=0;r<8;r++){
      int row = m0 + r + 8*hi;
      float v = acc[j][r] + bv;
      if (MODE == 0)      ((float*)outp)[(size_t)row*ldc + oc] += v;
      else if (MODE == 1) ((bf16*)outp)[(size_t)row*ldc + oc] = f2bf(v);
      else if (MODE == 2) ((bf16*)outp)[(size_t)row*ldc + oc] = f2bf(v > 0.f ? v : 0.f);
      else                ((bf16*)outp)[(size_t)row*ldc + oc] = f2bf(v);
    }
  }
}

// ---------------- embedding (multi-scale circular conv over electrodes) ------
struct EmbedW {
  const float *w1[5], *b1[5], *w3[5], *b3[5], *w5[5], *b5[5];
};
__global__ __launch_bounds__(256) void embed_kernel(const float* __restrict__ x,
                                                    EmbedW EW,
                                                    float* __restrict__ out)
{
  __shared__ float xs[5][DIN];     // 5 taps (superset for k=5)
  const int n = blockIdx.x;                 // token ((b*19+c)*128+t)
  const int b = n / (CC*TT);
  const int rem = n - b*(CC*TT);
  const int c = rem / TT;
  const int t = rem - c*TT;
  const int r  = (c < 16) ? (c >> 2) : 4;
  const int e  = (c < 16) ? (c & 3)  : (c - 16);
  const int ne = (r < 4) ? 4 : 3;
  const int s  = r * 4;
  const int tid = threadIdx.x;
  for (int idx = tid; idx < 5*DIN; idx += 256){
    int j = idx / DIN, i = idx - j*DIN;
    int ee = (e + j - 2 + 2*ne) % ne;       // circular over electrodes of the region
    xs[j][i] = x[(((size_t)(b*CC + s + ee))*TT + t)*DIN + i];
  }
  __syncthreads();
  const int d = tid;
  if (d >= DMOD) return;
  float acc;
  if (d < 100){
    acc = EW.b1[r][d];
    const float* wp = EW.w1[r] + (size_t)d * DIN;     // (100,200,1)
    for (int i=0;i<DIN;i++) acc += xs[2][i] * wp[i];
  } else if (d < 150){
    int dd = d - 100;
    acc = EW.b3[r][dd];
    const float* wp = EW.w3[r] + (size_t)dd * DIN * 3; // (50,200,3)
    for (int tap=0; tap<3; tap++)
      for (int i=0;i<DIN;i++) acc += xs[1+tap][i] * wp[i*3 + tap];
  } else {
    int dd = d - 150;
    acc = EW.b5[r][dd];
    const float* wp = EW.w5[r] + (size_t)dd * DIN * 5; // (50,200,5)
    for (int tap=0; tap<5; tap++)
      for (int i=0;i<DIN;i++) acc += xs[tap][i] * wp[i*5 + tap];
  }
  out[(size_t)n * DMOD + d] = acc;
}

// ---------------- LayerNorm -> bf16 (padded) ----------------
__global__ __launch_bounds__(256) void ln_to_bf16(const float* __restrict__ h,
                                                  const float* __restrict__ g,
                                                  const float* __restrict__ be,
                                                  bf16* __restrict__ out)
{
  const int w = threadIdx.x >> 5, lane = threadIdx.x & 31;
  const size_t tok = (size_t)blockIdx.x * 8 + w;
  const float* hp = h + tok * DMOD;
  float s = 0.f, sq = 0.f;
  for (int i = lane; i < DMOD; i += 32){ float v = hp[i]; s += v; sq += v*v; }
  for (int off = 16; off; off >>= 1){ s += __shfl_xor(s, off, 32); sq += __shfl_xor(sq, off, 32); }
  float mu = s * (1.f/DMOD);
  float var = sq * (1.f/DMOD) - mu*mu;
  float rs = rsqrtf(var + 1e-5f);
  bf16* op = out + tok * KP;
  for (int i = lane; i < DMOD; i += 32) op[i] = f2bf((hp[i]-mu)*rs*g[i] + be[i]);
  for (int i = DMOD + lane; i < KP; i += 32) op[i] = bfzero();
}

// ---------------- final LayerNorm (in place, fp32) ----------------
__global__ __launch_bounds__(256) void ln_final(float* __restrict__ h,
                                                const float* __restrict__ g,
                                                const float* __restrict__ be)
{
  const int w = threadIdx.x >> 5, lane = threadIdx.x & 31;
  const size_t tok = (size_t)blockIdx.x * 8 + w;
  float* hp = h + tok * DMOD;
  float s = 0.f, sq = 0.f;
  for (int i = lane; i < DMOD; i += 32){ float v = hp[i]; s += v; sq += v*v; }
  for (int off = 16; off; off >>= 1){ s += __shfl_xor(s, off, 32); sq += __shfl_xor(sq, off, 32); }
  float mu = s * (1.f/DMOD);
  float var = sq * (1.f/DMOD) - mu*mu;
  float rs = rsqrtf(var + 1e-5f);
  for (int i = lane; i < DMOD; i += 32) hp[i] = (hp[i]-mu)*rs*g[i] + be[i];
}

// ---------------- time attention (seq=128 over T, per channel) --------------
__global__ __launch_bounds__(256) void time_attn(const bf16* __restrict__ QKV,
                                                 bf16* __restrict__ Aout)
{
  __shared__ bf16 Ks[128*64];       // K rows as-is
  __shared__ bf16 VT[64*128];       // V transposed: [dim][key]
  __shared__ bf16 Pls[8*16*128];    // per-wave probability tiles
  const int bc = blockIdx.x >> 2, head = blockIdx.x & 3;
  const int tid = threadIdx.x, w = tid >> 5, lane = tid & 31;
  const int hi = lane >> 4, lm = lane & 15;
  const size_t base = (size_t)bc * TT * LDQKV;
  const bf16* Qg = QKV + base + head*64;
  const bf16* Kg = QKV + base + 256 + head*64;
  const bf16* Vg = QKV + base + 512 + head*64;
  for (int idx = tid; idx < 128*64; idx += 256){
    int key = idx >> 6, d = idx & 63;
    Ks[idx] = Kg[(size_t)key*LDQKV + d];
    VT[d*128 + key] = Vg[(size_t)key*LDQKV + d];
  }
  __syncthreads();
  const int q0 = w * 16;
  const bf16* qrow = Qg + (size_t)(q0 + lm) * LDQKV;
  bf16x16 qf0 = load_a(qrow, hi);
  bf16x16 qf1 = load_a(qrow + 32, hi);
  f32x8 s[8];
#pragma unroll
  for (int nt = 0; nt < 8; nt++){
    f32x8 acc = fz8();
    const bf16* krow = Ks + (nt*16 + lm)*64;
    acc = wmma_bf(qf0, load_b(krow,      hi), acc);
    acc = wmma_bf(qf1, load_b(krow + 32, hi), acc);
#pragma unroll
    for (int i=0;i<8;i++) s[nt][i] = acc[i] * SCALE_ATTN;
  }
  bf16* Pw = Pls + w * 2048;
#pragma unroll
  for (int r = 0; r < 8; r++){
    float m = s[0][r];
#pragma unroll
    for (int nt = 1; nt < 8; nt++) m = fmaxf(m, s[nt][r]);
    m = fmaxf(m, __shfl_xor(m, 1, 32));
    m = fmaxf(m, __shfl_xor(m, 2, 32));
    m = fmaxf(m, __shfl_xor(m, 4, 32));
    m = fmaxf(m, __shfl_xor(m, 8, 32));
    float p[8]; float e = 0.f;
#pragma unroll
    for (int nt = 0; nt < 8; nt++){ p[nt] = __expf(s[nt][r] - m); e += p[nt]; }
    e += __shfl_xor(e, 1, 32); e += __shfl_xor(e, 2, 32);
    e += __shfl_xor(e, 4, 32); e += __shfl_xor(e, 8, 32);
    float inv = 1.f / e;
    int row = r + 8*hi;
#pragma unroll
    for (int nt = 0; nt < 8; nt++) Pw[row*128 + nt*16 + lm] = f2bf(p[nt]*inv);
  }
#pragma unroll
  for (int dt = 0; dt < 4; dt++){
    f32x8 o = fz8();
#pragma unroll
    for (int kc = 0; kc < 128; kc += 32){
      bf16x16 af  = load_a(Pw + lm*128 + kc, hi);
      bf16x16 bfg = load_b(VT + (dt*16 + lm)*128 + kc, hi);
      o = wmma_bf(af, bfg, o);
    }
    int d = dt*16 + lm;
    if (d < DH){
#pragma unroll
      for (int r = 0; r < 8; r++){
        int trow = q0 + r + 8*hi;
        Aout[((size_t)(bc*TT + trow))*KP + head*DH + d] = f2bf(o[r]);
      }
    }
  }
}

// ---------------- region group means ----------------
__global__ __launch_bounds__(64) void group_mean(const bf16* __restrict__ A,
                                                 bf16* __restrict__ G)
{
  const int g = blockIdx.x;                 // (b*5+r)*128+t
  const int t = g & 127; const int br = g >> 7;
  const int r = br % 5;  const int b  = br / 5;
  const int s = r * 4;   const int ne = (r < 4) ? 4 : 3;
  const float inv = 1.f / (float)ne;
  bf16* gp = G + (size_t)g * KP;
  for (int col = threadIdx.x; col < KP; col += 64){
    if (col < DMOD){
      float acc = 0.f;
      for (int j = 0; j < ne; j++)
        acc += bf2f(A[(((size_t)(b*CC + s + j))*TT + t)*KP + col]);
      gp[col] = f2bf(acc * inv);
    } else gp[col] = bfzero();
  }
}

// ---------------- combine hf + gf -> bf16 ----------------
__global__ __launch_bounds__(256) void combine_hf_gf(const bf16* __restrict__ A,
                                                     const bf16* __restrict__ GF,
                                                     bf16* __restrict__ F)
{
  const int w = threadIdx.x >> 5, lane = threadIdx.x & 31;
  const size_t tok = (size_t)blockIdx.x * 8 + w;
  const int b = (int)(tok / (CC*TT));
  const int rem = (int)(tok - (size_t)b*(CC*TT));
  const int c = rem / TT; const int t = rem - c*TT;
  const int r = (c < 16) ? (c >> 2) : 4;
  const size_t grow = ((size_t)(b*5 + r)*TT + t) * KP;
  bf16* fp = F + tok * KP;
  const bf16* ap = A + tok * KP;
  for (int col = lane; col < KP; col += 32){
    if (col < DMOD) fp[col] = f2bf(bf2f(ap[col]) + bf2f(GF[grow + col]));
    else            fp[col] = bfzero();
  }
}

// ---------------- region attention (seq=19 over channels, masked) -----------
__global__ __launch_bounds__(128) void region_attn(const bf16* __restrict__ QKV,
                                                   bf16* __restrict__ Aout)
{
  const int bt = blockIdx.x;                // b*128+t
  const int b = bt >> 7, t = bt & 127;
  const int head = threadIdx.x >> 5;
  const int c = threadIdx.x & 31;
  if (c >= CC) return;
  const int r  = (c < 16) ? (c >> 2) : 4;
  const int e  = (c < 16) ? (c & 3)  : (c - 16);
  const int ne = (r < 4) ? 4 : 3;
  const size_t tok = ((size_t)(b*CC + c)*TT + t);
  const bf16* qp = QKV + tok*LDQKV + head*64;
  float q[DH];
#pragma unroll
  for (int i=0;i<DH;i++) q[i] = bf2f(qp[i]);
  float s[CC]; float mx = -1e30f;
  for (int j = 0; j < CC; j++){
    int rj  = (j < 16) ? (j >> 2) : 4;
    int ej  = (j < 16) ? (j & 3)  : (j - 16);
    int nej = (rj < 4) ? 4 : 3;
    bool same = false;
    for (int g = 0; g < 4; g++) if ((g % ne) == e && (g % nej) == ej) same = true;
    const size_t tj = ((size_t)(b*CC + j)*TT + t);
    const bf16* kp = QKV + tj*LDQKV + 256 + head*64;
    float acc = 0.f;
#pragma unroll
    for (int i=0;i<DH;i++) acc += q[i] * bf2f(kp[i]);
    acc *= SCALE_ATTN;
    if (!same) acc = -1e30f;
    s[j] = acc; mx = fmaxf(mx, acc);
  }
  float se = 0.f;
  for (int j = 0; j < CC; j++){ s[j] = __expf(s[j] - mx); se += s[j]; }
  float inv = 1.f / se;
  float o[DH];
#pragma unroll
  for (int i=0;i<DH;i++) o[i] = 0.f;
  for (int j = 0; j < CC; j++){
    float p = s[j] * inv;
    const size_t tj = ((size_t)(b*CC + j)*TT + t);
    const bf16* vp = QKV + tj*LDQKV + 512 + head*64;
#pragma unroll
    for (int i=0;i<DH;i++) o[i] += p * bf2f(vp[i]);
  }
  bf16* op = Aout + tok*KP + head*DH;
#pragma unroll
  for (int i=0;i<DH;i++) op[i] = f2bf(o[i]);
}

// =================== host side ===================
struct HostLayer {
  const float *ln1g,*ln1b,*ln2g,*ln2b,*ln3g,*ln3b;
  const float *win_in_w,*win_in_b,*win_out_w,*win_out_b;
  const float *wr_in_w,*wr_in_b,*wr_out_w,*wr_out_b;
  const float *gfc_w,*gfc_b,*ff1_w,*ff1_b,*ff2_w,*ff2_b;
};
struct LayW { bf16 *win_in,*win_out,*wr_in,*wr_out,*gfc,*ff1,*ff2; };

static void launch_gemm(int mode, const bf16* A, int lda, int kIters,
                        const bf16* W, int ldw, const float* bias, int Kout,
                        void* out, int ldc, int Mtiles, hipStream_t s)
{
  int KoutT = (Kout + 15) / 16;
  dim3 grid(Mtiles, (KoutT + 15) / 16);
  dim3 blk(128);
  switch (mode){
    case 0: gemm_wmma_k<0><<<grid, blk, 0, s>>>(A, lda, kIters, W, ldw, bias, Kout, out, ldc); break;
    case 1: gemm_wmma_k<1><<<grid, blk, 0, s>>>(A, lda, kIters, W, ldw, bias, Kout, out, ldc); break;
    case 2: gemm_wmma_k<2><<<grid, blk, 0, s>>>(A, lda, kIters, W, ldw, bias, Kout, out, ldc); break;
    default: gemm_wmma_k<3><<<grid, blk, 0, s>>>(A, lda, kIters, W, ldw, bias, Kout, out, ldc); break;
  }
}

extern "C" void kernel_launch(void* const* d_in, const int* in_sizes, int n_in,
                              void* d_out, int out_size, void* d_ws, size_t ws_size,
                              hipStream_t stream)
{
  auto g = [&](int i)->const float* { return (const float*)d_in[i]; };

  // ---- resolve input ordering (insertion vs jax-sorted pytree flatten) ----
  const float* x = nullptr;
  EmbedW EW{};
  HostLayer HL[2]{};
  const float *fg = nullptr, *fb = nullptr;

  bool x_first = (n_in > 0 && in_sizes[0] == NTOK*DIN/ (TT*CC) * (TT*CC)) && (in_sizes[0] == BB*CC*TT*DIN);
  if (x_first && in_sizes[1] == 20000){
    // Case A: insertion order throughout
    int i = 0; x = g(i++);
    for (int r=0;r<5;r++){ EW.w1[r]=g(i++); EW.b1[r]=g(i++); EW.w3[r]=g(i++); EW.b3[r]=g(i++); EW.w5[r]=g(i++); EW.b5[r]=g(i++); }
    for (int l=0;l<2;l++){ HostLayer& L=HL[l];
      L.ln1g=g(i++); L.ln1b=g(i++); L.ln2g=g(i++); L.ln2b=g(i++); L.ln3g=g(i++); L.ln3b=g(i++);
      L.win_in_w=g(i++); L.win_in_b=g(i++); L.win_out_w=g(i++); L.win_out_b=g(i++);
      L.wr_in_w=g(i++); L.wr_in_b=g(i++); L.wr_out_w=g(i++); L.wr_out_b=g(i++);
      L.gfc_w=g(i++); L.gfc_b=g(i++); L.ff1_w=g(i++); L.ff1_b=g(i++); L.ff2_w=g(i++); L.ff2_b=g(i++); }
    fg=g(i++); fb=g(i++);
  } else if (x_first){
    // Case B: x first, params flattened with sorted dict keys
    int i = 0; x = g(i++);
    for (int r=0;r<5;r++){ EW.b1[r]=g(i++); EW.b3[r]=g(i++); EW.b5[r]=g(i++); EW.w1[r]=g(i++); EW.w3[r]=g(i++); EW.w5[r]=g(i++); }
    fg=g(i++); fb=g(i++);
    for (int l=0;l<2;l++){ HostLayer& L=HL[l];
      L.ff1_b=g(i++); L.ff1_w=g(i++); L.ff2_b=g(i++); L.ff2_w=g(i++);
      L.gfc_b=g(i++); L.gfc_w=g(i++);
      L.ln1b=g(i++); L.ln1g=g(i++); L.ln2b=g(i++); L.ln2g=g(i++); L.ln3b=g(i++); L.ln3g=g(i++);
      L.win_in_b=g(i++); L.win_in_w=g(i++); L.win_out_b=g(i++); L.win_out_w=g(i++);
      L.wr_in_b=g(i++); L.wr_in_w=g(i++); L.wr_out_b=g(i++); L.wr_out_w=g(i++); }
  } else {
    // Case C: fully sorted ('params' < 'x'): params leaves then x
    int i = 0;
    for (int r=0;r<5;r++){ EW.b1[r]=g(i++); EW.b3[r]=g(i++); EW.b5[r]=g(i++); EW.w1[r]=g(i++); EW.w3[r]=g(i++); EW.w5[r]=g(i++); }
    fg=g(i++); fb=g(i++);
    for (int l=0;l<2;l++){ HostLayer& L=HL[l];
      L.ff1_b=g(i++); L.ff1_w=g(i++); L.ff2_b=g(i++); L.ff2_w=g(i++);
      L.gfc_b=g(i++); L.gfc_w=g(i++);
      L.ln1b=g(i++); L.ln1g=g(i++); L.ln2b=g(i++); L.ln2g=g(i++); L.ln3b=g(i++); L.ln3g=g(i++);
      L.win_in_b=g(i++); L.win_in_w=g(i++); L.win_out_b=g(i++); L.win_out_w=g(i++);
      L.wr_in_b=g(i++); L.wr_in_w=g(i++); L.wr_out_b=g(i++); L.wr_out_w=g(i++); }
    x = g(i++);
  }

  // ---- workspace carve ----
  size_t off = 0;
  auto alloc = [&](size_t elems)->bf16* {
    bf16* r = (bf16*)((char*)d_ws + off);
    off += ((elems*sizeof(bf16) + 255) & ~(size_t)255);
    return r;
  };
  bf16* Abuf = alloc((size_t)NTOK * KP);      // LN out / attention out
  bf16* Fbuf = alloc((size_t)NTOK * DFF);     // hf+gf (stride KP) / ff1 out (stride DFF)
  bf16* QKVb = alloc((size_t)NTOK * LDQKV);   // head-padded q|k|v
  bf16* Gbuf = alloc((size_t)NG * KP);        // group means
  bf16* GFb  = alloc((size_t)NG * KP);        // gfc output
  LayW LW[2];
  for (int l=0;l<2;l++){
    LW[l].win_in  = alloc((size_t)608*KP);
    LW[l].win_out = alloc((size_t)208*KP);
    LW[l].wr_in   = alloc((size_t)608*KP);
    LW[l].wr_out  = alloc((size_t)208*KP);
    LW[l].gfc     = alloc((size_t)208*KP);
    LW[l].ff1     = alloc((size_t)DFF*KP);
    LW[l].ff2     = alloc((size_t)208*DFF);
  }
  (void)ws_size;
  hipMemsetAsync(d_ws, 0, off, stream);       // zero padding once per call

  // ---- convert weights to padded bf16 ----
  auto cvt = [&](const float* src, int rows, int cols, bf16* dst, int ldd){
    int tot = rows * cols;
    cvt_w_kernel<<<(tot + 255)/256, 256, 0, stream>>>(src, rows, cols, dst, ldd);
  };
  for (int l=0;l<2;l++){
    cvt(HL[l].win_in_w, 600, 200, LW[l].win_in, KP);
    cvt(HL[l].win_out_w,200, 200, LW[l].win_out,KP);
    cvt(HL[l].wr_in_w,  600, 200, LW[l].wr_in,  KP);
    cvt(HL[l].wr_out_w, 200, 200, LW[l].wr_out, KP);
    cvt(HL[l].gfc_w,    200, 200, LW[l].gfc,    KP);
    cvt(HL[l].ff1_w,    800, 200, LW[l].ff1,    KP);
    cvt(HL[l].ff2_w,    200, 800, LW[l].ff2,    DFF);
  }

  float* resid = (float*)d_out;               // residual stream lives in d_out
  (void)out_size;

  // ---- embedding ----
  embed_kernel<<<NTOK, 256, 0, stream>>>(x, EW, resid);

  const int MT = NTOK / 16;                   // 4864 token tiles
  // ---- encoder layers ----
  for (int l=0;l<2;l++){
    const HostLayer& L = HL[l]; const LayW& W = LW[l];
    // -- inter-window (time) attention --
    ln_to_bf16<<<NTOK/8, 256, 0, stream>>>(resid, L.ln1g, L.ln1b, Abuf);
    launch_gemm(3, Abuf, KP, KP/32, W.win_in, KP, L.win_in_b, 600, QKVb, LDQKV, MT, stream);
    time_attn<<<(BB*CC)*NHEAD, 256, 0, stream>>>(QKVb, Abuf);
    launch_gemm(0, Abuf, KP, KP/32, W.win_out, KP, L.win_out_b, 200, resid, DMOD, MT, stream);
    // -- inter-region attention --
    ln_to_bf16<<<NTOK/8, 256, 0, stream>>>(resid, L.ln2g, L.ln2b, Abuf);
    group_mean<<<NG, 64, 0, stream>>>(Abuf, Gbuf);
    launch_gemm(1, Gbuf, KP, KP/32, W.gfc, KP, L.gfc_b, 200, GFb, KP, NG/16, stream);
    combine_hf_gf<<<NTOK/8, 256, 0, stream>>>(Abuf, GFb, Fbuf);
    launch_gemm(3, Fbuf, KP, KP/32, W.wr_in, KP, L.wr_in_b, 600, QKVb, LDQKV, MT, stream);
    region_attn<<<BB*TT, 128, 0, stream>>>(QKVb, Abuf);
    launch_gemm(0, Abuf, KP, KP/32, W.wr_out, KP, L.wr_out_b, 200, resid, DMOD, MT, stream);
    // -- feed forward --
    ln_to_bf16<<<NTOK/8, 256, 0, stream>>>(resid, L.ln3g, L.ln3b, Abuf);
    launch_gemm(2, Abuf, KP, KP/32, W.ff1, KP, L.ff1_b, 800, Fbuf, DFF, MT, stream);
    launch_gemm(0, Fbuf, DFF, DFF/32, W.ff2, DFF, L.ff2_b, 200, resid, DMOD, MT, stream);
  }
  // ---- final layer norm (in place) ----
  ln_final<<<NTOK/8, 256, 0, stream>>>(resid, fg, fb);
}
